// Decoder_5050881540099
// MI455X (gfx1250) — matrix operand order repaired and checked by reading, not compiled, not run on hardware
//
#include <hip/hip_runtime.h>

// ---------------------------------------------------------------------------
// GraphCast-style decoder for MI455X (gfx1250, wave32, WMMA).
// v2: 32 rows per block, B-fragment reuse across two row-tiles (2 accumulators
// per B load) -> halves L2 weight traffic, doubles wmma per vmem load.
// Exploits: dst features are all-zero (edge_proc K: 768->512), latlon x rows
// are zero (node_proc K: 512->256, residual vanishes), edge_dst groups of 7
// are contiguous (low-contention f32 atomics into L2-resident agg).
// GEMMs: v_wmma_f32_16x16x32_bf16, fp32 accumulate; LN/SiLU/bias fp32.
// ---------------------------------------------------------------------------

#define NUM_LATLON 16200
#define NUM_H3     5882
#define E_TOT      (7 * NUM_LATLON)   // 113400
#define FDIM       256
#define OUT_DIM    78
#define LN_EPS     1e-5f

#define TR  32    // rows per block
#define LDA 264   // LDS row stride (elements); 528B rows: 16B aligned, bank-friendly

typedef __attribute__((ext_vector_type(16))) __bf16 v16bf;
typedef __attribute__((ext_vector_type(8)))  __bf16 v8bf;
typedef __attribute__((ext_vector_type(8)))  float  v8f;

static __device__ __forceinline__ __bf16 f2bf(float f) {
  unsigned u = __builtin_bit_cast(unsigned, f);
  u += 0x7FFFu + ((u >> 16) & 1u);             // round-to-nearest-even
  unsigned short h = (unsigned short)(u >> 16);
  return __builtin_bit_cast(__bf16, h);
}

static __device__ __forceinline__ float silu_f(float v) {
  return v / (1.f + __expf(-v));
}

// A fragment: 16x32 bf16 tile from LDS activation buffer rows [m0..m0+15].
// Lane l: row M=l&15; halves 0..7 = K(k0 + hi*8 ..), halves 8..15 = K(+16).
static __device__ __forceinline__ v16bf load_a(const __bf16* buf, int k0, int lane) {
  int m = lane & 15, hi = lane >> 4;
  const __bf16* p = buf + m * LDA + k0 + hi * 8;
  v8bf lo = *(const v8bf*)p;            // 16B LDS load
  v8bf hp = *(const v8bf*)(p + 16);     // 16B LDS load
  return __builtin_shufflevector(lo, hp, 0,1,2,3,4,5,6,7,8,9,10,11,12,13,14,15);
}

// B fragment: 32x16 bf16 tile from transposed weights Wt[N][ldk].
// Lane l: col N=n0+(l&15); halves 0..7 = K(k0 + hi*8 ..), halves 8..15 = K(+16).
static __device__ __forceinline__ v16bf load_b(const __bf16* wt, int ldk, int n0, int k0, int lane) {
  int n = n0 + (lane & 15), hi = lane >> 4;
  const __bf16* p = wt + (long long)n * ldk + k0 + hi * 8;
  v8bf lo = *(const v8bf*)p;
  v8bf hp = *(const v8bf*)(p + 16);
  return __builtin_shufflevector(lo, hp, 0,1,2,3,4,5,6,7,8,9,10,11,12,13,14,15);
}

#define WMMA_BF16(acc, a, b) \
  __builtin_amdgcn_wmma_f32_16x16x32_bf16(false, (a), false, (b), (short)0, (acc), false, false)

// Store one 16x16 fp32 accumulator tile with bias (+optional SiLU+bf16).
static __device__ __forceinline__ void store_tile_bf(v8f acc, __bf16* dst, int n, int hi, float bn) {
#pragma unroll
  for (int j = 0; j < 8; ++j)
    dst[(j + hi * 8) * LDA + n] = f2bf(silu_f(acc[j] + bn));
}
static __device__ __forceinline__ void store_tile_f32(v8f acc, float* dst, int n, int hi, float bn) {
#pragma unroll
  for (int j = 0; j < 8; ++j)
    dst[(j + hi * 8) * LDA + n] = acc[j] + bn;
}

// Fused layer over TR=32 rows: dst = act(src[32xK] @ W[KxN] + bias).
// 4 waves x 16-col tiles; each B fragment is reused for 2 row-tile WMMAs.
static __device__ __forceinline__ void gemm_layer(
    const __bf16* __restrict__ src, const __bf16* __restrict__ wt, int ldk, int kk,
    const float* __restrict__ bias, int N, int lane, int wave,
    __bf16* dstb, float* dstf)
{
  int ntiles = N >> 4;
  for (int t = wave; t < ntiles; t += 4) {
    int n0 = t << 4;
    v8f acc0 = {}, acc1 = {};
    for (int k0 = 0; k0 < kk; k0 += 32) {
      v16bf bfrag = load_b(wt, ldk, n0, k0, lane);
      acc0 = WMMA_BF16(acc0, load_a(src, k0, lane), bfrag);
      acc1 = WMMA_BF16(acc1, load_a(src + 16 * LDA, k0, lane), bfrag);
    }
    int n = n0 + (lane & 15), hi = lane >> 4;
    float bn = bias[n];
    if (dstb) {
      store_tile_bf(acc0, dstb, n, hi, bn);
      store_tile_bf(acc1, dstb + 16 * LDA, n, hi, bn);
    } else {
      store_tile_f32(acc0, dstf, n, hi, bn);
      store_tile_f32(acc1, dstf + 16 * LDA, n, hi, bn);
    }
  }
}

// LayerNorm over 256 cols of a 32-row fp32 LDS tile [32][LDA], in place.
// red: >= 320 floats scratch. Ends with __syncthreads().
static __device__ __forceinline__ void layernorm32(
    float* f, const float* __restrict__ g, const float* __restrict__ be,
    float* red, int tid)
{
  int row = tid >> 2, part = tid & 3;          // 4 threads per row, 64 cols each
  float* fr = f + row * LDA + part * 64;
  float s = 0.f, s2 = 0.f;
#pragma unroll 8
  for (int i = 0; i < 64; ++i) { float x = fr[i]; s += x; s2 += x * x; }
  red[row * 8 + part] = s;
  red[row * 8 + 4 + part] = s2;
  __syncthreads();
  if (tid < 32) {
    float ss = 0.f, q = 0.f;
#pragma unroll
    for (int p = 0; p < 4; ++p) { ss += red[tid * 8 + p]; q += red[tid * 8 + 4 + p]; }
    float mu = ss * (1.f / 256.f);
    float var = q * (1.f / 256.f) - mu * mu;
    red[256 + tid * 2] = mu;
    red[256 + tid * 2 + 1] = rsqrtf(var + LN_EPS);
  }
  __syncthreads();
  float mu = red[256 + row * 2], rstd = red[256 + row * 2 + 1];
#pragma unroll 8
  for (int i = 0; i < 64; ++i) {
    int c = part * 64 + i;
    fr[i] = (fr[i] - mu) * rstd * g[c] + be[c];
  }
  __syncthreads();
}

// Pack fp32 weight [K][N] -> bf16 transposed [Npad][K] (zero pad cols >= N).
__global__ void pack_t_kernel(const float* __restrict__ src, __bf16* __restrict__ dst,
                              int K, int N, int Npad)
{
  int i = blockIdx.x * blockDim.x + threadIdx.x;
  if (i >= K * Npad) return;
  int n = i / K, k = i - n * K;
  dst[(long long)n * K + k] = (n < N) ? f2bf(src[(long long)k * N + n]) : f2bf(0.f);
}

// ------------------- fused edge pipeline: edge_enc + edge_proc + scatter ----
__global__ __launch_bounds__(128)
void edge_kernel(const float* __restrict__ pf, const int* __restrict__ esrc,
                 const int* __restrict__ edst, const float* __restrict__ efeat,
                 const float* eeW0, const float* eeB0, const float* eeB1, const float* eeB2,
                 const float* eeG, const float* eeBe,
                 const float* epB0, const float* epB1, const float* epB2,
                 const float* epG, const float* epBe,
                 const __bf16* W1E, const __bf16* W2E,
                 const __bf16* WP0, const __bf16* WP1, const __bf16* WP2,
                 float* __restrict__ agg)
{
  __shared__ __align__(16) __bf16 act0[TR * LDA];
  __shared__ __align__(16) __bf16 act1[TR * LDA];
  __shared__ __align__(16) __bf16 act2[TR * LDA];
  __shared__ __align__(16) float  f32a[TR * LDA];   // LN staging (e0, then e)
  __shared__ float red[320];
  __shared__ float efe[TR * 2];
  __shared__ int   sidx[TR], didx[TR];

  int tid = threadIdx.x, lane = tid & 31, wave = tid >> 5;
  int b = blockIdx.y;
  int ebase = blockIdx.x * TR;

  if (tid < TR) {
    int e = ebase + tid; if (e > E_TOT - 1) e = E_TOT - 1;
    efe[tid * 2 + 0] = efeat[e * 2 + 0];
    efe[tid * 2 + 1] = efeat[e * 2 + 1];
    sidx[tid] = esrc[e];
    didx[tid] = edst[e] - NUM_H3;
  }
  __syncthreads();

  // edge_enc L0: 2 -> 256, SiLU (tiny; plain VALU)
  for (int i = tid; i < TR * 256; i += 128) {
    int r = i >> 8, c = i & 255;
    float v = efe[r * 2] * eeW0[c] + efe[r * 2 + 1] * eeW0[256 + c] + eeB0[c];
    act0[r * LDA + c] = f2bf(silu_f(v));
  }
  __syncthreads();

  gemm_layer(act0, W1E, 256, 256, eeB1, 256, lane, wave, act1, nullptr);  // L1 SiLU
  __syncthreads();
  gemm_layer(act1, W2E, 256, 256, eeB2, 256, lane, wave, nullptr, f32a);  // L2 plain
  __syncthreads();
  layernorm32(f32a, eeG, eeBe, red, tid);                                 // e0 fp32

  // bf16 copy of e0 -> act0 (WMMA input + residual); gather src rows -> act1
  for (int i = tid; i < TR * 256; i += 128) {
    int r = i >> 8, c = i & 255;
    act0[r * LDA + c] = f2bf(f32a[r * LDA + c]);
    act1[r * LDA + c] = f2bf(pf[((long long)(b * NUM_H3 + sidx[r])) * FDIM + c]);
  }
  __syncthreads();

  // edge_proc L1: K=512 (src uses W rows 0..255, e0 uses W rows 512..767)
  for (int t = wave; t < 16; t += 4) {
    int n0 = t << 4;
    v8f acc0 = {}, acc1 = {};
    for (int k0 = 0; k0 < 256; k0 += 32) {
      v16bf bfrag = load_b(WP0, 768, n0, k0, lane);
      acc0 = WMMA_BF16(acc0, load_a(act1, k0, lane), bfrag);
      acc1 = WMMA_BF16(acc1, load_a(act1 + 16 * LDA, k0, lane), bfrag);
    }
    for (int k0 = 0; k0 < 256; k0 += 32) {
      v16bf bfrag = load_b(WP0, 768, n0, 512 + k0, lane);
      acc0 = WMMA_BF16(acc0, load_a(act0, k0, lane), bfrag);
      acc1 = WMMA_BF16(acc1, load_a(act0 + 16 * LDA, k0, lane), bfrag);
    }
    int n = n0 + (lane & 15), hi = lane >> 4;
    float bn = epB0[n];
    store_tile_bf(acc0, act2, n, hi, bn);
    store_tile_bf(acc1, act2 + 16 * LDA, n, hi, bn);
  }
  __syncthreads();
  gemm_layer(act2, WP1, 256, 256, epB1, 256, lane, wave, act1, nullptr);  // L2 SiLU
  __syncthreads();
  gemm_layer(act1, WP2, 256, 256, epB2, 256, lane, wave, nullptr, f32a);  // L3 plain
  __syncthreads();
  layernorm32(f32a, epG, epBe, red, tid);

  // e = LN(..) + e0 ; scatter-add into agg[b][dst][:]
  for (int i = tid; i < TR * 256; i += 128) {
    int r = i >> 8, c = i & 255;
    if (ebase + r < E_TOT) {
      float v = f32a[r * LDA + c] + (float)act0[r * LDA + c];
      atomicAdd(agg + ((long long)b * NUM_LATLON + didx[r]) * FDIM + c, v);
    }
  }
}

// ------------------- fused node pipeline: node_proc + node_dec + output -----
__global__ __launch_bounds__(128)
void node_kernel(const float* __restrict__ agg, const float* __restrict__ startf,
                 const float* npB0, const float* npB1, const float* npB2,
                 const float* npG, const float* npBe,
                 const float* ndB0, const float* ndB1, const float* ndB2,
                 const __bf16* WN0, const __bf16* WN1, const __bf16* WN2,
                 const __bf16* WD0, const __bf16* WD1, const __bf16* WD2,
                 float* __restrict__ out)
{
  __shared__ __align__(16) __bf16 act0[TR * LDA];
  __shared__ __align__(16) __bf16 act1[TR * LDA];
  __shared__ __align__(16) __bf16 act2[TR * LDA];
  __shared__ __align__(16) float  f32a[TR * LDA];
  __shared__ float red[320];

  int tid = threadIdx.x, lane = tid & 31, wave = tid >> 5;
  int b = blockIdx.y;
  int r0 = blockIdx.x * TR;

  for (int i = tid; i < TR * 256; i += 128) {
    int r = i >> 8, c = i & 255;
    int row = r0 + r; if (row > NUM_LATLON - 1) row = NUM_LATLON - 1;
    act0[r * LDA + c] = f2bf(agg[((long long)b * NUM_LATLON + row) * FDIM + c]);
  }
  __syncthreads();

  // node_proc L1: x-part is zero -> only W rows 256..511, K=256
  for (int t = wave; t < 16; t += 4) {
    int n0 = t << 4;
    v8f acc0 = {}, acc1 = {};
    for (int k0 = 0; k0 < 256; k0 += 32) {
      v16bf bfrag = load_b(WN0, 512, n0, 256 + k0, lane);
      acc0 = WMMA_BF16(acc0, load_a(act0, k0, lane), bfrag);
      acc1 = WMMA_BF16(acc1, load_a(act0 + 16 * LDA, k0, lane), bfrag);
    }
    int n = n0 + (lane & 15), hi = lane >> 4;
    float bn = npB0[n];
    store_tile_bf(acc0, act1, n, hi, bn);
    store_tile_bf(acc1, act1 + 16 * LDA, n, hi, bn);
  }
  __syncthreads();
  gemm_layer(act1, WN1, 256, 256, npB1, 256, lane, wave, act2, nullptr);
  __syncthreads();
  gemm_layer(act2, WN2, 256, 256, npB2, 256, lane, wave, nullptr, f32a);
  __syncthreads();
  layernorm32(f32a, npG, npBe, red, tid);          // h (residual +x = 0 here)

  for (int i = tid; i < TR * 256; i += 128) {
    int r = i >> 8, c = i & 255;
    act0[r * LDA + c] = f2bf(f32a[r * LDA + c]);
  }
  __syncthreads();

  gemm_layer(act0, WD0, 256, 256, ndB0, 128, lane, wave, act1, nullptr);  // 256->128 SiLU
  __syncthreads();
  gemm_layer(act1, WD1, 128, 128, ndB1, 128, lane, wave, act2, nullptr);  // 128->128 SiLU
  __syncthreads();

  // final 128->78 (weights padded to 80 cols) + start_features, direct store
  for (int t = wave; t < 5; t += 4) {
    int n0 = t << 4;
    v8f acc0 = {}, acc1 = {};
    for (int k0 = 0; k0 < 128; k0 += 32) {
      v16bf bfrag = load_b(WD2, 128, n0, k0, lane);
      acc0 = WMMA_BF16(acc0, load_a(act2, k0, lane), bfrag);
      acc1 = WMMA_BF16(acc1, load_a(act2 + 16 * LDA, k0, lane), bfrag);
    }
    int n = n0 + (lane & 15), hi = lane >> 4;
    if (n < OUT_DIM) {
      float bn = ndB2[n];
#pragma unroll
      for (int j = 0; j < 8; ++j) {
        int row = r0 + j + hi * 8;
        if (row < NUM_LATLON) {
          long long o = ((long long)b * NUM_LATLON + row) * OUT_DIM + n;
          out[o] = acc0[j] + bn + startf[o];
        }
        int row2 = row + 16;
        if (row2 < NUM_LATLON) {
          long long o2 = ((long long)b * NUM_LATLON + row2) * OUT_DIM + n;
          out[o2] = acc1[j] + bn + startf[o2];
        }
      }
    }
  }
}

// ---------------------------------------------------------------------------
extern "C" void kernel_launch(void* const* d_in, const int* in_sizes, int n_in,
                              void* d_out, int out_size, void* d_ws, size_t ws_size,
                              hipStream_t stream) {
  // Inputs: setup_inputs() dict order; params pytree flattened (dict keys
  // alphabetical: edge_enc, edge_proc, node_dec, node_proc; each b, ln, w).
  const float* pf     = (const float*)d_in[0];
  const float* startf = (const float*)d_in[1];
  const int*   esrc   = (const int*)d_in[2];
  const int*   edst   = (const int*)d_in[3];
  const float* efeat  = (const float*)d_in[4];
  const float* eeB0 = (const float*)d_in[5],  *eeB1 = (const float*)d_in[6],
             * eeB2 = (const float*)d_in[7],  *eeG  = (const float*)d_in[8],
             * eeBe = (const float*)d_in[9],  *eeW0 = (const float*)d_in[10],
             * eeW1 = (const float*)d_in[11], *eeW2 = (const float*)d_in[12];
  const float* epB0 = (const float*)d_in[13], *epB1 = (const float*)d_in[14],
             * epB2 = (const float*)d_in[15], *epG  = (const float*)d_in[16],
             * epBe = (const float*)d_in[17], *epW0 = (const float*)d_in[18],
             * epW1 = (const float*)d_in[19], *epW2 = (const float*)d_in[20];
  const float* ndB0 = (const float*)d_in[21], *ndB1 = (const float*)d_in[22],
             * ndB2 = (const float*)d_in[23], *ndW0 = (const float*)d_in[24],
             * ndW1 = (const float*)d_in[25], *ndW2 = (const float*)d_in[26];
  const float* npB0 = (const float*)d_in[27], *npB1 = (const float*)d_in[28],
             * npB2 = (const float*)d_in[29], *npG  = (const float*)d_in[30],
             * npBe = (const float*)d_in[31], *npW0 = (const float*)d_in[32],
             * npW1 = (const float*)d_in[33], *npW2 = (const float*)d_in[34];

  // Workspace: [agg fp32 (2,16200,256)] then packed bf16 transposed weights.
  const size_t AGG_BYTES = (size_t)2 * NUM_LATLON * FDIM * sizeof(float);
  float*  agg = (float*)d_ws;
  __bf16* wb  = (__bf16*)((char*)d_ws + AGG_BYTES);
  __bf16* W1E = wb;             // [256][256]
  __bf16* W2E = W1E + 65536;    // [256][256]
  __bf16* WP0 = W2E + 65536;    // [256][768]
  __bf16* WP1 = WP0 + 196608;   // [256][256]
  __bf16* WP2 = WP1 + 65536;    // [256][256]
  __bf16* WN0 = WP2 + 65536;    // [256][512]
  __bf16* WN1 = WN0 + 131072;   // [256][256]
  __bf16* WN2 = WN1 + 65536;    // [256][256]
  __bf16* WD0 = WN2 + 65536;    // [128][256]
  __bf16* WD1 = WD0 + 32768;    // [128][128]
  __bf16* WD2 = WD1 + 16384;    // [80][128] (padded from 78)

  hipMemsetAsync(agg, 0, AGG_BYTES, stream);

  auto pk = [&](const float* s, __bf16* d, int K, int N, int Npad) {
    int total = K * Npad;
    pack_t_kernel<<<(total + 255) / 256, 256, 0, stream>>>(s, d, K, N, Npad);
  };
  pk(eeW1, W1E, 256, 256, 256);
  pk(eeW2, W2E, 256, 256, 256);
  pk(epW0, WP0, 768, 256, 256);
  pk(epW1, WP1, 256, 256, 256);
  pk(epW2, WP2, 256, 256, 256);
  pk(npW0, WN0, 512, 256, 256);
  pk(npW1, WN1, 256, 256, 256);
  pk(npW2, WN2, 256, 256, 256);
  pk(ndW0, WD0, 256, 128, 128);
  pk(ndW1, WD1, 128, 128, 128);
  pk(ndW2, WD2, 128,  78,  80);

  dim3 eg((E_TOT + TR - 1) / TR, 2);
  edge_kernel<<<eg, 128, 0, stream>>>(pf, esrc, edst, efeat,
      eeW0, eeB0, eeB1, eeB2, eeG, eeBe,
      epB0, epB1, epB2, epG, epBe,
      W1E, W2E, WP0, WP1, WP2, agg);

  dim3 ng((NUM_LATLON + TR - 1) / TR, 2);
  node_kernel<<<ng, 128, 0, stream>>>(agg, startf,
      npB0, npB1, npB2, npG, npBe,
      ndB0, ndB1, ndB2,
      WN0, WN1, WN2, WD0, WD1, WD2, (float*)d_out);
}
